// LSTM_80582176407707
// MI455X (gfx1250) — matrix operand order repaired and checked by reading, not compiled, not run on hardware
//
#include <hip/hip_runtime.h>
#include <hip/hip_bf16.h>
#include <math.h>
#include <stdint.h>

// ---------------------------------------------------------------------------
// 2-layer LSTM, B=32, T=2048, I=256, H=512, O=256.
// Persistent-kernel design for MI455X (gfx1250, wave32, WMMA):
//   * recurrent weights live in LDS (fp16) for the whole scan,
//     staged once via the Tensor Data Mover (tensor_load_to_lds)
//   * per-step h broadcast via GLOBAL_LOAD_ASYNC_TO_LDS_B128 (ASYNCcnt)
//   * 32 workgroups, each owns a 16-wide hidden slice of both layers
//   * v_wmma_f32_16x16x32_f16 for all matmuls, fp32 accumulate
//   * 2 grid barriers per timestep (atomic phase barrier in workspace)
// ---------------------------------------------------------------------------

#define B_  32
#define T_  2048
#define I_  256
#define H_  512
#define G_  2048   // 4*H
#define O_  256
#define NWG 32

typedef __attribute__((ext_vector_type(16))) _Float16 v16h;
typedef __attribute__((ext_vector_type(8)))  _Float16 v8h;
typedef __attribute__((ext_vector_type(8)))  float    v8f;
typedef uint32_t u32x4 __attribute__((ext_vector_type(4)));
typedef uint32_t u32x8 __attribute__((ext_vector_type(8)));

// ---- workspace layout (bytes) ----
#define OFF_XT    ((size_t)0)                          // fp16 x, [T][B][I]
#define OFF_WIH0  (OFF_XT   + (size_t)T_*B_*I_*2)      // fp16 [2048][256]
#define OFF_WHH0  (OFF_WIH0 + (size_t)G_*I_*2)         // fp16 [2048][512]
#define OFF_WIH1  (OFF_WHH0 + (size_t)G_*H_*2)
#define OFF_WHH1  (OFF_WIH1 + (size_t)G_*H_*2)
#define OFF_FCW   (OFF_WHH1 + (size_t)G_*H_*2)         // fp16 [256][512]
#define OFF_B0    (OFF_FCW  + (size_t)O_*H_*2)         // f32 [2048]
#define OFF_B1    (OFF_B0   + (size_t)G_*4)
#define OFF_H0G   (OFF_B1   + (size_t)G_*4)            // fp16 [32][512]
#define OFF_H1G   (OFF_H0G  + (size_t)B_*H_*2)
#define OFF_BAR   (OFF_H1G  + (size_t)B_*H_*2)         // barrier counter
#define WS_NEED   (OFF_BAR + 256)
#define ZERO_OFF  OFF_H0G
#define ZERO_LEN  (WS_NEED - OFF_H0G)

// ---- LDS layout (bytes), total 307200 < 320KB ----
#define LDS_WIH0  0          // 64*256*2 = 32768
#define LDS_WHH0  32768      // 64*512*2 = 65536
#define LDS_WIH1  98304
#define LDS_WHH1  163840
#define LDS_H0    229376     // 32*512*2 = 32768
#define LDS_H1    262144
#define LDS_SCR   294912     // 8*16*16*4 = 8192
#define LDS_C0    303104     // 32*16*4
#define LDS_C1    305152
#define LDS_TOTAL 307200

// ---------------- WMMA fragment helpers (contiguous k order, same in A and B) --------------------
// A (16x32 f16): lane<16 -> M=lane, K=k0..k0+15
//                lane>=16 -> M=lane-16, K=k0+16..k0+31
__device__ __forceinline__ v16h load_a_frag(const _Float16* A, int lda,
                                            int m0, int k0, int lane) {
  const int m  = m0 + (lane & 15);
  const int kb = (lane >> 4) * 16;
  const _Float16* p = A + (size_t)m * lda + k0 + kb;
  return *(const v16h*)(p);
}

// B (32x16 f16) from row-major W[n][k] (i.e. B = W^T):
// lane<16 -> N=lane, K=k0..k0+15 ; lane>=16 -> N=lane-16, K=k0+16..k0+31
__device__ __forceinline__ v16h load_b_frag(const _Float16* W, int ldw,
                                            int n0, int k0, int lane) {
  const int n  = n0 + (lane & 15);
  const int kb = (lane >> 4) * 16;
  return *(const v16h*)(W + (size_t)n * ldw + k0 + kb);
}

__device__ __forceinline__ v8f wmma_f16(v16h a, v16h b, v8f c) {
  return __builtin_amdgcn_wmma_f32_16x16x32_f16(false, a, false, b,
                                                (short)0, c, false, false);
}

__device__ __forceinline__ float sigm(float x) {
  return 1.0f / (1.0f + __expf(-x));
}

// ---------------- CDNA5 async / TDM data movers ----------------------------
// One instruction copies 16B per lane (512B per wave), memory -> LDS,
// tracked by ASYNCcnt.  lds/g addresses are per-lane.
__device__ __forceinline__ void async_g2l_b128(uint32_t lds_addr,
                                               const void* gaddr) {
  asm volatile("global_load_async_to_lds_b128 %0, %1, off"
               :: "v"(lds_addr), "v"(gaddr) : "memory");
}
__device__ __forceinline__ void wait_async0() {
  asm volatile("s_wait_asynccnt 0x0" ::: "memory");
}

// copy 32KB global -> LDS with 8 async b128 ops per wave
__device__ __forceinline__ void async_copy32k(void* lds, const void* g,
                                              int tid) {
  const uint32_t l = (uint32_t)(uintptr_t)lds;   // low 32b of flat = LDS addr
  const char* s = (const char*)g;
#pragma unroll
  for (int j = 0; j < 8; ++j) {
    const int i = tid + j * 256;
    async_g2l_b128(l + i * 16, s + (size_t)i * 16);
  }
}

// Tensor Data Mover: 1-D tile of `nelem` fp16 elements, global -> LDS.
// D# per CDNA5 ISA 8.3/8.4: count=1, type=2, data_size=2B,
// tensor_dim0 = tile_dim0 = nelem, tensor_dim1 = tile_dim1 = 1.
__device__ __forceinline__ void tdm_load_1d(uint32_t lds_addr,
                                            const void* gsrc,
                                            uint32_t nelem) {
  const uint64_t ga = (uint64_t)(uintptr_t)gsrc;
  u32x4 d0;
  d0[0] = 1u;                                         // count=1 (valid)
  d0[1] = lds_addr;                                   // lds_addr
  d0[2] = (uint32_t)ga;                               // global_addr[31:0]
  d0[3] = ((uint32_t)(ga >> 32) & 0x01FFFFFFu)        // global_addr[56:32]
          | 0x80000000u;                              // type=2 (bits 127:126)
  u32x8 d1;
  d1[0] = 0x00010000u;                                // data_size=1 (2 bytes)
  d1[1] = (nelem & 0xFFFFu) << 16;                    // tensor_dim0[15:0]
  d1[2] = ((nelem >> 16) & 0xFFFFu) | (1u << 16);     // dim0 hi | tensor_dim1=1
  d1[3] = (nelem & 0xFFFFu) << 16;                    // tile_dim0
  d1[4] = 1u;                                         // tile_dim1=1, tile_dim2=0
  d1[5] = nelem;                                      // tensor_dim0_stride lo
  d1[6] = 0u;
  d1[7] = 0u;
  asm volatile("tensor_load_to_lds %0, %1" :: "s"(d0), "s"(d1) : "memory");
}
__device__ __forceinline__ void wait_tensor0() {
  asm volatile("s_wait_tensorcnt 0x0" ::: "memory");
}

// atomic phase grid barrier (all NWG blocks resident -> no deadlock)
__device__ __forceinline__ void gbar(int* arrive, int& phase) {
  __syncthreads();
  ++phase;
  if (threadIdx.x == 0) {
    __threadfence();
    atomicAdd(arrive, 1);
    const int target = phase * NWG;
    while (__hip_atomic_load(arrive, __ATOMIC_RELAXED,
                             __HIP_MEMORY_SCOPE_AGENT) < target)
      __builtin_amdgcn_s_sleep(1);
    __threadfence();
  }
  __syncthreads();
}

// ---------------------------- prep kernels ---------------------------------
__global__ void cvt_f32_to_f16(_Float16* __restrict__ dst,
                               const float* __restrict__ src, int n) {
  int i = blockIdx.x * 256 + threadIdx.x;
  if (i < n) dst[i] = (_Float16)src[i];
}

__global__ void bias_sum(float* __restrict__ dst, const float* __restrict__ a,
                         const float* __restrict__ b, int n) {
  int i = blockIdx.x * 256 + threadIdx.x;
  if (i < n) dst[i] = a[i] + b[i];
}

// x[b][t][i] (f32) -> xT[t][b][i] (f16)
__global__ void xpose_cvt(_Float16* __restrict__ xT,
                          const float* __restrict__ x) {
  int id = blockIdx.x * 256 + threadIdx.x;   // == (t*32 + b)*256 + i
  int i = id & 255;
  int b = (id >> 8) & 31;
  int t = id >> 13;
  xT[id] = (_Float16)x[((size_t)b * T_ + t) * I_ + i];
}

// ------------------------- persistent LSTM kernel --------------------------
__global__ void __launch_bounds__(256, 1)
lstm_persistent(const _Float16* __restrict__ xT,
                const _Float16* __restrict__ Wih0,
                const _Float16* __restrict__ Whh0,
                const _Float16* __restrict__ Wih1,
                const _Float16* __restrict__ Whh1,
                const float* __restrict__ bias0,
                const float* __restrict__ bias1,
                _Float16* __restrict__ h0g,
                _Float16* __restrict__ h1g,
                int* arrive) {
  extern __shared__ char smem[];
  _Float16* Wih0_s = (_Float16*)(smem + LDS_WIH0);
  _Float16* Whh0_s = (_Float16*)(smem + LDS_WHH0);
  _Float16* Wih1_s = (_Float16*)(smem + LDS_WIH1);
  _Float16* Whh1_s = (_Float16*)(smem + LDS_WHH1);
  _Float16* hbuf0  = (_Float16*)(smem + LDS_H0);
  _Float16* hbuf1  = (_Float16*)(smem + LDS_H1);
  float*    scr    = (float*)(smem + LDS_SCR);
  float*    c0     = (float*)(smem + LDS_C0);
  float*    c1     = (float*)(smem + LDS_C1);

  const int tid  = threadIdx.x;
  const int w16  = blockIdx.x * 16;     // this WG's hidden-column slice
  const int lane = tid & 31;
  const int wave = tid >> 5;            // 8 waves
  const int rt   = wave & 1;            // batch row-tile (0: b0-15, 1: b16-31)
  const int g    = wave >> 1;           // gate index i/f/g/o
  const int m0   = rt * 16;
  const int scro = ((rt << 2) + g) << 8;  // scr tile base = (rt*4+g)*256

  // ---- stage weight slices into LDS once via the Tensor Data Mover ----
  // For each matrix, gate q's 16-row slice (rows q*512+w16 .. +15) is a
  // contiguous 16*K-element block in the row-major [4H][K] weights.
  if (tid < 32) {                       // wave 0 issues all TDM descriptors
#pragma unroll
    for (int q = 0; q < 4; ++q) {
      const size_t grow = (size_t)(q * 512 + w16);
      tdm_load_1d((uint32_t)(uintptr_t)Wih0_s + q * 16 * I_ * 2,
                  Wih0 + grow * I_, 16 * I_);
      tdm_load_1d((uint32_t)(uintptr_t)Whh0_s + q * 16 * H_ * 2,
                  Whh0 + grow * H_, 16 * H_);
      tdm_load_1d((uint32_t)(uintptr_t)Wih1_s + q * 16 * H_ * 2,
                  Wih1 + grow * H_, 16 * H_);
      tdm_load_1d((uint32_t)(uintptr_t)Whh1_s + q * 16 * H_ * 2,
                  Whh1 + grow * H_, 16 * H_);
    }
    wait_tensor0();
  }
  for (int e = tid; e < B_ * 16; e += 256) { c0[e] = 0.f; c1[e] = 0.f; }
  __syncthreads();

  int phase = 0;

  for (int t = 0; t < T_; ++t) {
    // ---- broadcast-load h0(t-1), h1(t-1) into LDS (async, no VGPR hop) ----
    async_copy32k(hbuf0, h0g, tid);
    async_copy32k(hbuf1, h1g, tid);
    wait_async0();
    __syncthreads();

    // ================= layer 0: gates = x_t*Wih0^T + h0*Whh0^T =============
    {
      const _Float16* xt = xT + (size_t)t * (B_ * I_);
      // warm L2/WGP$ for next timestep's x tile while we compute
      if (t + 1 < T_) __builtin_prefetch(xt + B_ * I_ + tid * 32, 0, 0);
      v8f acc = {};
#pragma unroll
      for (int k0 = 0; k0 < I_; k0 += 32)
        acc = wmma_f16(load_a_frag(xt, I_, m0, k0, lane),
                       load_b_frag(Wih0_s, I_, g * 16, k0, lane), acc);
#pragma unroll
      for (int k0 = 0; k0 < H_; k0 += 32)
        acc = wmma_f16(load_a_frag(hbuf0, H_, m0, k0, lane),
                       load_b_frag(Whh0_s, H_, g * 16, k0, lane), acc);
      float* s = scr + scro;
#pragma unroll
      for (int v = 0; v < 8; ++v)
        s[(((lane >> 4) << 3) + v) * 16 + (lane & 15)] = acc[v];
    }
    __syncthreads();

    // ---- layer-0 elementwise update of our 16 hidden columns ----
    for (int e = tid; e < B_ * 16; e += 256) {
      int b = e >> 4, j = e & 15;
      int brt = b >> 4, m = b & 15;
      int o = (brt << 10) + m * 16 + j;
      float ig = sigm(scr[o + 0 * 256] + bias0[0 * H_ + w16 + j]);
      float fg = sigm(scr[o + 1 * 256] + bias0[1 * H_ + w16 + j]);
      float gg = tanhf(scr[o + 2 * 256] + bias0[2 * H_ + w16 + j]);
      float og = sigm(scr[o + 3 * 256] + bias0[3 * H_ + w16 + j]);
      float c = fg * c0[e] + ig * gg;
      c0[e] = c;
      h0g[b * H_ + w16 + j] = (_Float16)(og * tanhf(c));
    }
    gbar(arrive, phase);   // h0(t) now globally visible

    // reload full h0(t) (async)
    async_copy32k(hbuf0, h0g, tid);
    wait_async0();
    __syncthreads();

    // ================= layer 1: gates = h0*Wih1^T + h1*Whh1^T ==============
    {
      v8f acc = {};
#pragma unroll
      for (int k0 = 0; k0 < H_; k0 += 32)
        acc = wmma_f16(load_a_frag(hbuf0, H_, m0, k0, lane),
                       load_b_frag(Wih1_s, H_, g * 16, k0, lane), acc);
#pragma unroll
      for (int k0 = 0; k0 < H_; k0 += 32)
        acc = wmma_f16(load_a_frag(hbuf1, H_, m0, k0, lane),
                       load_b_frag(Whh1_s, H_, g * 16, k0, lane), acc);
      float* s = scr + scro;
#pragma unroll
      for (int v = 0; v < 8; ++v)
        s[(((lane >> 4) << 3) + v) * 16 + (lane & 15)] = acc[v];
    }
    __syncthreads();

    for (int e = tid; e < B_ * 16; e += 256) {
      int b = e >> 4, j = e & 15;
      int brt = b >> 4, m = b & 15;
      int o = (brt << 10) + m * 16 + j;
      float ig = sigm(scr[o + 0 * 256] + bias1[0 * H_ + w16 + j]);
      float fg = sigm(scr[o + 1 * 256] + bias1[1 * H_ + w16 + j]);
      float gg = tanhf(scr[o + 2 * 256] + bias1[2 * H_ + w16 + j]);
      float og = sigm(scr[o + 3 * 256] + bias1[3 * H_ + w16 + j]);
      float c = fg * c1[e] + ig * gg;
      c1[e] = c;
      h1g[b * H_ + w16 + j] = (_Float16)(og * tanhf(c));
    }
    gbar(arrive, phase);   // h1(t) now globally visible
  }
}

// --------------------------- final FC (WMMA) -------------------------------
__global__ void __launch_bounds__(256)
fc_kernel(const _Float16* __restrict__ h1g, const _Float16* __restrict__ fcw,
          const float* __restrict__ fcb, float* __restrict__ out) {
  const int lane = threadIdx.x & 31;
  const int wv   = threadIdx.x >> 5;
  for (int ct = wv; ct < O_ / 16; ct += 8) {
#pragma unroll
    for (int rt = 0; rt < 2; ++rt) {
      v8f acc = {};
#pragma unroll
      for (int k0 = 0; k0 < H_; k0 += 32)
        acc = wmma_f16(load_a_frag(h1g, H_, rt * 16, k0, lane),
                       load_b_frag(fcw, H_, ct * 16, k0, lane), acc);
      const int n = ct * 16 + (lane & 15);
#pragma unroll
      for (int v = 0; v < 8; ++v) {
        int m = rt * 16 + ((lane >> 4) << 3) + v;
        out[m * O_ + n] = acc[v] + fcb[n];
      }
    }
  }
}

// ------------------------------- launcher ----------------------------------
extern "C" void kernel_launch(void* const* d_in, const int* in_sizes, int n_in,
                              void* d_out, int out_size, void* d_ws,
                              size_t ws_size, hipStream_t stream) {
  (void)in_sizes; (void)n_in; (void)out_size;
  if (ws_size < WS_NEED) return;

  const float* x     = (const float*)d_in[0];
  const float* Wih0f = (const float*)d_in[1];
  const float* Whh0f = (const float*)d_in[2];
  const float* bih0  = (const float*)d_in[3];
  const float* bhh0  = (const float*)d_in[4];
  const float* Wih1f = (const float*)d_in[5];
  const float* Whh1f = (const float*)d_in[6];
  const float* bih1  = (const float*)d_in[7];
  const float* bhh1  = (const float*)d_in[8];
  const float* fcwf  = (const float*)d_in[9];
  const float* fcb   = (const float*)d_in[10];

  char* ws = (char*)d_ws;
  _Float16* xT    = (_Float16*)(ws + OFF_XT);
  _Float16* Wih0h = (_Float16*)(ws + OFF_WIH0);
  _Float16* Whh0h = (_Float16*)(ws + OFF_WHH0);
  _Float16* Wih1h = (_Float16*)(ws + OFF_WIH1);
  _Float16* Whh1h = (_Float16*)(ws + OFF_WHH1);
  _Float16* fcwh  = (_Float16*)(ws + OFF_FCW);
  float*    b0    = (float*)(ws + OFF_B0);
  float*    b1    = (float*)(ws + OFF_B1);
  _Float16* h0g   = (_Float16*)(ws + OFF_H0G);
  _Float16* h1g   = (_Float16*)(ws + OFF_H1G);
  int*      bar   = (int*)(ws + OFF_BAR);

  // zero h0, h1, barrier (deterministic per launch / graph replay)
  hipMemsetAsync(ws + ZERO_OFF, 0, ZERO_LEN, stream);

  cvt_f32_to_f16<<<(G_ * I_ + 255) / 256, 256, 0, stream>>>(Wih0h, Wih0f, G_ * I_);
  cvt_f32_to_f16<<<(G_ * H_ + 255) / 256, 256, 0, stream>>>(Whh0h, Whh0f, G_ * H_);
  cvt_f32_to_f16<<<(G_ * H_ + 255) / 256, 256, 0, stream>>>(Wih1h, Wih1f, G_ * H_);
  cvt_f32_to_f16<<<(G_ * H_ + 255) / 256, 256, 0, stream>>>(Whh1h, Whh1f, G_ * H_);
  cvt_f32_to_f16<<<(O_ * H_ + 255) / 256, 256, 0, stream>>>(fcwh, fcwf, O_ * H_);
  bias_sum<<<(G_ + 255) / 256, 256, 0, stream>>>(b0, bih0, bhh0, G_);
  bias_sum<<<(G_ + 255) / 256, 256, 0, stream>>>(b1, bih1, bhh1, G_);
  xpose_cvt<<<(T_ * B_ * I_) / 256, 256, 0, stream>>>(xT, x);

  hipFuncSetAttribute((const void*)lstm_persistent,
                      hipFuncAttributeMaxDynamicSharedMemorySize, LDS_TOTAL);
  lstm_persistent<<<NWG, 256, LDS_TOTAL, stream>>>(
      xT, Wih0h, Whh0h, Wih1h, Whh1h, b0, b1, h0g, h1g, bar);

  fc_kernel<<<1, 256, 0, stream>>>(h1g, fcwh, fcb, (float*)d_out);
}
